// Cross_Attention_63702954934333
// MI455X (gfx1250) — matrix operand order repaired and checked by reading, not compiled, not run on hardware
//
#include <hip/hip_runtime.h>
#include <hip/hip_bf16.h>

// ---------------------------------------------------------------------------
// Cross-attention block for MI455X (gfx1250, wave32, WMMA).
// All GEMMs: v_wmma_f32_16x16x32_bf16 (fp32 accumulate).
// ---------------------------------------------------------------------------

typedef __bf16 bf16_t;
typedef __attribute__((ext_vector_type(16))) __bf16 v16bf;
typedef __attribute__((ext_vector_type(8)))  __bf16 v8bf;
typedef __attribute__((ext_vector_type(8)))  float  v8f;

#define BB 4
#define NN 1024
#define CC 512
#define HH 8
#define HD 64
#define ATTN_SCALE 0.125f   // HD^-0.5

__device__ __forceinline__ v8f wmma_bf16(v16bf a, v16bf b, v8f c) {
  return __builtin_amdgcn_wmma_f32_16x16x32_bf16(
      /*neg_a=*/false, a, /*neg_b=*/false, b,
      /*c_mod=*/(short)0, c, /*reuse_a=*/false, /*reuse_b=*/false);
}

// A fragment, 16-bit 16x32 (MxK). Row-major source, leading dim `ld`.
// Order used here (contiguous k, same as B): lane L%16 = M; lanes 0-15 hold K 0..15, lanes 16-31 hold K 16..31.
__device__ __forceinline__ v16bf load_A(const bf16_t* __restrict__ base, int ld,
                                        int m0, int k0) {
  const int lane = threadIdx.x & 31;
  const bf16_t* p = base + (m0 + (lane & 15)) * ld + k0 + ((lane >> 4) << 4);
  v16bf a;
#pragma unroll
  for (int j = 0; j < 8; ++j) {
    const int kb = 2 * j;
    a[2 * j]     = p[kb];
    a[2 * j + 1] = p[kb + 1];
  }
  return a;
}

// Same, with per-row fp32 scale (for normalized-q Gram matrix).
__device__ __forceinline__ v16bf load_A_scaled(const bf16_t* __restrict__ base,
                                               const float* __restrict__ inv,
                                               int ld, int m0, int k0) {
  const int lane = threadIdx.x & 31;
  const int row = m0 + (lane & 15);
  const float s = inv[row];
  const bf16_t* p = base + row * ld + k0 + ((lane >> 4) << 4);
  v16bf a;
#pragma unroll
  for (int j = 0; j < 8; ++j) {
    const int kb = 2 * j;
    a[2 * j]     = (bf16_t)((float)p[kb] * s);
    a[2 * j + 1] = (bf16_t)((float)p[kb + 1] * s);
  }
  return a;
}

// B fragment (32x16 KxN) where B = Mt^T and Mt is row-major [n, k] (ld = k dim).
// Order used here (contiguous k, same as A): lane L%16 = N; lanes 0-15 hold K 0..15, lanes 16-31 hold K 16..31.
__device__ __forceinline__ v16bf load_Bt(const bf16_t* __restrict__ mat, int ld,
                                         int n0, int k0) {
  const int lane = threadIdx.x & 31;
  const bf16_t* p = mat + (n0 + (lane & 15)) * ld + k0 + ((lane >> 4) << 4);
  v16bf b;
#pragma unroll
  for (int j = 0; j < 8; ++j) {
    b[2 * j]     = p[2 * j];
    b[2 * j + 1] = p[2 * j + 1];
  }
  return b;
}

__device__ __forceinline__ v16bf load_Bt_scaled(const bf16_t* __restrict__ mat,
                                                const float* __restrict__ inv,
                                                int ld, int n0, int k0) {
  const int lane = threadIdx.x & 31;
  const int n = n0 + (lane & 15);
  const float s = inv[n];
  const bf16_t* p = mat + n * ld + k0 + ((lane >> 4) << 4);
  v16bf b;
#pragma unroll
  for (int j = 0; j < 8; ++j) {
    b[2 * j]     = (bf16_t)((float)p[2 * j] * s);
    b[2 * j + 1] = (bf16_t)((float)p[2 * j + 1] * s);
  }
  return b;
}

// B fragment from row-major [k, n] source (ld = n dim)  — used for P @ V.
__device__ __forceinline__ v16bf load_Brm(const bf16_t* __restrict__ mat, int ld,
                                          int k0, int n0) {
  const int lane = threadIdx.x & 31;
  const bf16_t* p = mat + (k0 + ((lane >> 4) << 4)) * ld + n0 + (lane & 15);
  v16bf b;
#pragma unroll
  for (int j = 0; j < 8; ++j) {
    b[2 * j]     = p[(2 * j) * ld];
    b[2 * j + 1] = p[(2 * j + 1) * ld];
  }
  return b;
}

// ---------------------------------------------------------------------------
// Kernels
// ---------------------------------------------------------------------------

__global__ void cast_f32_to_bf16(const float* __restrict__ in,
                                 bf16_t* __restrict__ out, int n) {
  int i = blockIdx.x * blockDim.x + threadIdx.x;
  if (i < n) out[i] = (bf16_t)in[i];
}

__global__ void zero_scalars(float* __restrict__ p) {
  if (threadIdx.x < 3) p[threadIdx.x] = 0.0f;
}

// QKV projection: [B*N, C] @ [4C, C]^T, scattered into [4, B, H, N, HD] bf16.
__global__ void __launch_bounds__(256)
qkv_gemm(const bf16_t* __restrict__ X, const bf16_t* __restrict__ W,
         bf16_t* __restrict__ qkv) {
  const int tilesN = (4 * CC) / 16;                     // 128
  const int total  = (BB * NN / 16) * tilesN;           // 32768 wave-tiles
  const int wid = (blockIdx.x * blockDim.x + threadIdx.x) >> 5;
  if (wid >= total) return;                             // wave-uniform
  const int m0 = (wid / tilesN) * 16;
  const int n0 = (wid % tilesN) * 16;
  v8f c = {};
#pragma unroll 4
  for (int k0 = 0; k0 < CC; k0 += 32)
    c = wmma_bf16(load_A(X, CC, m0, k0), load_Bt(W, CC, n0, k0), c);
  const int lane = threadIdx.x & 31;
  const int col = n0 + (lane & 15);
  const int chunk = col >> 9, rem = col & 511;
  const int h = rem >> 6, d = rem & 63;
#pragma unroll
  for (int v = 0; v < 8; ++v) {
    const int row = m0 + v + ((lane >> 4) << 3);
    const int bi = row >> 10, n = row & 1023;
    qkv[(((size_t)(chunk * BB + bi) * HH + h) * NN + n) * HD + d] = (bf16_t)c[v];
  }
}

// inv[row] = 1 / max(||t[row,:HD]||, eps)   (F.normalize denominator)
__global__ void row_invnorm(const bf16_t* __restrict__ t,
                            float* __restrict__ inv, int rows) {
  int r = blockIdx.x * blockDim.x + threadIdx.x;
  if (r >= rows) return;
  const bf16_t* p = t + (size_t)r * HD;
  float s = 0.0f;
#pragma unroll 8
  for (int i = 0; i < HD; ++i) { float f = (float)p[i]; s += f * f; }
  inv[r] = 1.0f / fmaxf(sqrtf(s), 1e-12f);
}

// sum over off-diagonal of (Qn @ Qtn^T)^2, scaled, atomically accumulated.
__global__ void __launch_bounds__(256)
ortho_loss(const bf16_t* __restrict__ q, const bf16_t* __restrict__ qt,
           const float* __restrict__ invq, const float* __restrict__ invqt,
           float* __restrict__ out, float scale) {
  __shared__ float wsum[8];
  const int tilesBH = (NN / 16) * (NN / 16);            // 4096
  const int wid = (blockIdx.x * blockDim.x + threadIdx.x) >> 5;
  const int bh = wid / tilesBH;
  const int tt = wid % tilesBH;
  const int i0 = (tt >> 6) << 4;
  const int j0 = (tt & 63) << 4;
  const bf16_t* qb  = q  + (size_t)bh * NN * HD;
  const bf16_t* qtb = qt + (size_t)bh * NN * HD;
  const float*  iq  = invq  + (size_t)bh * NN;
  const float*  iqt = invqt + (size_t)bh * NN;
  v8f c = {};
#pragma unroll
  for (int k0 = 0; k0 < HD; k0 += 32)
    c = wmma_bf16(load_A_scaled(qb, iq, HD, i0, k0),
                  load_Bt_scaled(qtb, iqt, HD, j0, k0), c);
  const int lane = threadIdx.x & 31;
  const int col = j0 + (lane & 15);
  float acc = 0.0f;
#pragma unroll
  for (int v = 0; v < 8; ++v) {
    const int row = i0 + v + ((lane >> 4) << 3);
    const float val = c[v];
    if (row != col) acc += val * val;
  }
  for (int off = 16; off > 0; off >>= 1) acc += __shfl_down(acc, off, 32);
  if (lane == 0) wsum[threadIdx.x >> 5] = acc;
  __syncthreads();
  if (threadIdx.x == 0) {
    float s = 0.0f;
    for (int w = 0; w < 8; ++w) s += wsum[w];
    atomicAdd(out, s * scale);
  }
}

// One wave per (b,h,16-row tile): QK^T -> LDS bf16 scores -> softmax -> P@V.
// bf16 score strip = 32 KB LDS -> ~10 resident waves per WGP (320 KB budget),
// and the P A-fragments become two contiguous 16-byte LDS reads per lane.
__global__ void __launch_bounds__(32)
attn_kernel(const bf16_t* __restrict__ q, const bf16_t* __restrict__ k,
            const bf16_t* __restrict__ v, bf16_t* __restrict__ out) {
  __shared__ __align__(16) bf16_t scb[16 * NN];         // 32 KB
  const int t  = blockIdx.x;
  const int bh = t >> 6;
  const int m0 = (t & 63) << 4;
  const bf16_t* qb = q + (size_t)bh * NN * HD;
  const bf16_t* kb = k + (size_t)bh * NN * HD;
  const bf16_t* vb = v + (size_t)bh * NN * HD;
  const int lane = threadIdx.x & 31;

  const v16bf aq0 = load_A(qb, HD, m0, 0);
  const v16bf aq1 = load_A(qb, HD, m0, 32);

  // Phase 1: scores = (Q K^T) * scale  -> LDS (bf16)
  for (int j = 0; j < NN / 16; ++j) {
    __builtin_prefetch(kb + (size_t)(j + 1) * 16 * HD, 0, 0);  // global_prefetch_b8
    v8f c = {};
    c = wmma_bf16(aq0, load_Bt(kb, HD, j * 16, 0), c);
    c = wmma_bf16(aq1, load_Bt(kb, HD, j * 16, 32), c);
    const int col = j * 16 + (lane & 15);
#pragma unroll
    for (int vv = 0; vv < 8; ++vv)
      scb[(vv + ((lane >> 4) << 3)) * NN + col] = (bf16_t)(c[vv] * ATTN_SCALE);
  }
  __syncthreads();

  // Phase 2: softmax, all 32 lanes: lane handles half a row, halves combined
  // with __shfl_xor across the lane-16 boundary.
  {
    bf16_t* sr = scb + (lane & 15) * NN + ((lane >> 4) * (NN / 2));
    float m = -3.4e38f;
    for (int i = 0; i < NN / 2; ++i) m = fmaxf(m, (float)sr[i]);
    m = fmaxf(m, __shfl_xor(m, 16, 32));
    float s = 0.0f;
    for (int i = 0; i < NN / 2; ++i) {
      const float e = __expf((float)sr[i] - m);
      sr[i] = (bf16_t)e;
      s += e;
    }
    s += __shfl_xor(s, 16, 32);
    const float r = 1.0f / s;
    for (int i = 0; i < NN / 2; ++i) sr[i] = (bf16_t)((float)sr[i] * r);
  }
  __syncthreads();

  // Phase 3: O = P @ V  (K = 1024 in chunks of 32); A fragment = 2x b128 LDS reads
  v8f o0 = {}, o1 = {}, o2 = {}, o3 = {};
  const bf16_t* sp = scb + (lane & 15) * NN;
  const int khalf = (lane >> 4) << 4;
  for (int ch = 0; ch < NN / 32; ++ch) {
    const int kb0 = ch * 32 + khalf;
    const v8bf lo = *(const v8bf*)(sp + kb0);            // K kb0 .. kb0+7
    const v8bf hi = *(const v8bf*)(sp + kb0 + 8);        // K kb0+8 .. kb0+15
    v16bf ap;
#pragma unroll
    for (int j = 0; j < 8; ++j) { ap[j] = lo[j]; ap[8 + j] = hi[j]; }
    o0 = wmma_bf16(ap, load_Brm(vb, HD, ch * 32, 0),  o0);
    o1 = wmma_bf16(ap, load_Brm(vb, HD, ch * 32, 16), o1);
    o2 = wmma_bf16(ap, load_Brm(vb, HD, ch * 32, 32), o2);
    o3 = wmma_bf16(ap, load_Brm(vb, HD, ch * 32, 48), o3);
  }

  const int bi = bh / HH, h = bh % HH;
  bf16_t* ob = out + (size_t)bi * NN * CC + h * HD;
#pragma unroll
  for (int vv = 0; vv < 8; ++vv) {
    const int row = m0 + vv + ((lane >> 4) << 3);
    const int d = lane & 15;
    ob[(size_t)row * CC + d]      = (bf16_t)o0[vv];
    ob[(size_t)row * CC + 16 + d] = (bf16_t)o1[vv];
    ob[(size_t)row * CC + 32 + d] = (bf16_t)o2[vv];
    ob[(size_t)row * CC + 48 + d] = (bf16_t)o3[vv];
  }
}

// out = X @ W^T + bias  (fp32 result straight into d_out)
__global__ void __launch_bounds__(256)
proj_gemm(const bf16_t* __restrict__ X, const bf16_t* __restrict__ W,
          const float* __restrict__ bias, float* __restrict__ out) {
  const int tilesN = CC / 16;                           // 32
  const int wid = (blockIdx.x * blockDim.x + threadIdx.x) >> 5;
  const int m0 = (wid / tilesN) * 16;
  const int n0 = (wid % tilesN) * 16;
  v8f c = {};
#pragma unroll 4
  for (int k0 = 0; k0 < CC; k0 += 32)
    c = wmma_bf16(load_A(X, CC, m0, k0), load_Bt(W, CC, n0, k0), c);
  const int lane = threadIdx.x & 31;
  const int col = n0 + (lane & 15);
  const float bv = bias[col];
#pragma unroll
  for (int v = 0; v < 8; ++v) {
    const int row = m0 + v + ((lane >> 4) << 3);
    out[(size_t)row * CC + col] = c[v] + bv;
  }
}

// mean(1 - cos(xc, yc)) with F.normalize semantics, one wave per row.
__global__ void __launch_bounds__(256)
cos_loss(const float* __restrict__ xc, const float* __restrict__ yc,
         float* __restrict__ out, float scale) {
  __shared__ float wsum[8];
  const int wid = (blockIdx.x * blockDim.x + threadIdx.x) >> 5;
  const int lane = threadIdx.x & 31;
  const float* xr = xc + (size_t)wid * CC;
  const float* yr = yc + (size_t)wid * CC;
  float dot = 0.0f, nx = 0.0f, ny = 0.0f;
  for (int i = lane; i < CC; i += 32) {
    const float a = xr[i], b = yr[i];
    dot += a * b; nx += a * a; ny += b * b;
  }
  for (int off = 16; off > 0; off >>= 1) {
    dot += __shfl_down(dot, off, 32);
    nx  += __shfl_down(nx,  off, 32);
    ny  += __shfl_down(ny,  off, 32);
  }
  if (lane == 0) {
    const float cs =
        dot / (fmaxf(sqrtf(nx), 1e-12f) * fmaxf(sqrtf(ny), 1e-12f));
    wsum[threadIdx.x >> 5] = 1.0f - cs;
  }
  __syncthreads();
  if (threadIdx.x == 0) {
    float s = 0.0f;
    for (int w = 0; w < 8; ++w) s += wsum[w];
    atomicAdd(out, s * scale);
  }
}

// ---------------------------------------------------------------------------
// Host launch
// ---------------------------------------------------------------------------

extern "C" void kernel_launch(void* const* d_in, const int* in_sizes, int n_in,
                              void* d_out, int out_size, void* d_ws, size_t ws_size,
                              hipStream_t stream) {
  (void)in_sizes; (void)n_in; (void)out_size; (void)ws_size;
  const float* x     = (const float*)d_in[0];
  const float* y     = (const float*)d_in[1];
  const float* Wqkv1 = (const float*)d_in[2];
  const float* Wqkv2 = (const float*)d_in[3];
  const float* W1 = (const float*)d_in[4];  const float* b1 = (const float*)d_in[5];
  const float* W2 = (const float*)d_in[6];  const float* b2 = (const float*)d_in[7];
  const float* W3 = (const float*)d_in[8];  const float* b3 = (const float*)d_in[9];
  const float* W4 = (const float*)d_in[10]; const float* b4 = (const float*)d_in[11];
  float* out = (float*)d_out;

  const size_t BNC = (size_t)BB * NN * CC;     // 2097152
  const size_t W4C = (size_t)4 * CC * CC;      // 1048576
  const size_t WC  = (size_t)CC * CC;          // 262144
  const size_t BHN = (size_t)BB * HH * NN;     // 32768

  char* ws = (char*)d_ws;
  size_t off = 0;
  auto alloc = [&](size_t bytes) -> char* {
    char* p = ws + off;
    off = (off + bytes + 255) & ~(size_t)255;
    return p;
  };
  bf16_t* xb   = (bf16_t*)alloc(BNC * 2);
  bf16_t* yb   = (bf16_t*)alloc(BNC * 2);
  bf16_t* wq1b = (bf16_t*)alloc(W4C * 2);
  bf16_t* wq2b = (bf16_t*)alloc(W4C * 2);
  bf16_t* w1b  = (bf16_t*)alloc(WC * 2);
  bf16_t* w2b  = (bf16_t*)alloc(WC * 2);
  bf16_t* w3b  = (bf16_t*)alloc(WC * 2);
  bf16_t* w4b  = (bf16_t*)alloc(WC * 2);
  bf16_t* qkv1 = (bf16_t*)alloc(4 * BNC * 2); // [4,B,H,N,HD]
  bf16_t* qkv2 = (bf16_t*)alloc(4 * BNC * 2);
  bf16_t* a_yc = (bf16_t*)alloc(BNC * 2);     // attention outputs (pre-proj)
  bf16_t* a_xc = (bf16_t*)alloc(BNC * 2);
  bf16_t* a_xt = (bf16_t*)alloc(BNC * 2);
  bf16_t* a_yt = (bf16_t*)alloc(BNC * 2);
  float* invqx  = (float*)alloc(BHN * 4);
  float* invqtx = (float*)alloc(BHN * 4);
  float* invqy  = (float*)alloc(BHN * 4);
  float* invqty = (float*)alloc(BHN * 4);

  float* scalars = out + 4 * BNC;             // lorthx, lorthy, lc
  zero_scalars<<<1, 32, 0, stream>>>(scalars);

  auto cast = [&](const float* src, bf16_t* dst, size_t n) {
    cast_f32_to_bf16<<<(int)((n + 255) / 256), 256, 0, stream>>>(src, dst, (int)n);
  };
  cast(x, xb, BNC);      cast(y, yb, BNC);
  cast(Wqkv1, wq1b, W4C); cast(Wqkv2, wq2b, W4C);
  cast(W1, w1b, WC); cast(W2, w2b, WC); cast(W3, w3b, WC); cast(W4, w4b, WC);

  // QKV projections: 32768 wave-tiles each, 8 waves/block
  qkv_gemm<<<4096, 256, 0, stream>>>(xb, wq1b, qkv1);
  qkv_gemm<<<4096, 256, 0, stream>>>(yb, wq2b, qkv2);

  // Inverse row norms for q / qt of both streams
  const bf16_t* qx  = qkv1;               const bf16_t* qtx = qkv1 + 3 * BNC;
  const bf16_t* kx  = qkv1 + 1 * BNC;     const bf16_t* vx  = qkv1 + 2 * BNC;
  const bf16_t* qy  = qkv2;               const bf16_t* qty = qkv2 + 3 * BNC;
  const bf16_t* ky  = qkv2 + 1 * BNC;     const bf16_t* vy  = qkv2 + 2 * BNC;
  row_invnorm<<<(int)(BHN / 256), 256, 0, stream>>>(qx,  invqx,  (int)BHN);
  row_invnorm<<<(int)(BHN / 256), 256, 0, stream>>>(qtx, invqtx, (int)BHN);
  row_invnorm<<<(int)(BHN / 256), 256, 0, stream>>>(qy,  invqy,  (int)BHN);
  row_invnorm<<<(int)(BHN / 256), 256, 0, stream>>>(qty, invqty, (int)BHN);

  // Ortho losses: B*H*(N/16)^2 = 131072 wave-tiles / 8 per block
  const float oscale = 1.0f / ((float)BB * NN * NN);
  ortho_loss<<<16384, 256, 0, stream>>>(qx, qtx, invqx, invqtx, scalars + 0, oscale);
  ortho_loss<<<16384, 256, 0, stream>>>(qy, qty, invqy, invqty, scalars + 1, oscale);

  // Attention: B*H*(N/16) = 2048 single-wave workgroups (32 KB LDS each)
  attn_kernel<<<2048, 32, 0, stream>>>(qx,  ky, vy, a_yc);   // yc
  attn_kernel<<<2048, 32, 0, stream>>>(qy,  kx, vx, a_xc);   // xc
  attn_kernel<<<2048, 32, 0, stream>>>(qtx, kx, vx, a_xt);   // xt
  attn_kernel<<<2048, 32, 0, stream>>>(qty, ky, vy, a_yt);   // yt

  // Output projections -> d_out (xc, yc, xt, yt) as fp32
  proj_gemm<<<1024, 256, 0, stream>>>(a_xc, w1b, b1, out + 0 * BNC);
  proj_gemm<<<1024, 256, 0, stream>>>(a_yc, w2b, b2, out + 1 * BNC);
  proj_gemm<<<1024, 256, 0, stream>>>(a_xt, w3b, b3, out + 2 * BNC);
  proj_gemm<<<1024, 256, 0, stream>>>(a_yt, w4b, b4, out + 3 * BNC);

  // Cosine distance loss over B*N = 4096 rows, 8 waves/block
  cos_loss<<<512, 256, 0, stream>>>(out + 0 * BNC, out + 1 * BNC,
                                    scalars + 2, 1.0f / ((float)BB * NN));
}
